// DeepseekV2Layer_48198122995783
// MI455X (gfx1250) — matrix operand order repaired and checked by reading, not compiled, not run on hardware
//
#include <hip/hip_runtime.h>
#include <hip/hip_bf16.h>

// ---------------------------------------------------------------------------
// Types / helpers
// ---------------------------------------------------------------------------
typedef __attribute__((ext_vector_type(16))) __bf16 v16bf;
typedef __attribute__((ext_vector_type(8)))  __bf16 v8bf;
typedef __attribute__((ext_vector_type(8)))  float  v8f;
typedef __attribute__((ext_vector_type(4)))  unsigned v4u;
typedef __attribute__((ext_vector_type(8)))  int    v8i;
typedef __attribute__((ext_vector_type(4)))  int    v4i;

#define T_SEQ   1024
#define DMODEL  4096
#define NHEAD   32
#define D_NOPE  128
#define D_ROPE  64
#define D_HEAD  192
#define D_V     128
#define Q_RANK  1536
#define KV_RANK 512
#define I_FF    11008
#define ATTN_SCALE 0.072168783648703220563643597562744f  // 192^-0.5

__device__ __forceinline__ __bf16 f2bf(float f) { return (__bf16)f; }   // v_cvt_*_bf16_f32, RNE
__device__ __forceinline__ float bf2f(__bf16 b) { return (float)b; }

__device__ __forceinline__ v16bf combine16(v8bf lo, v8bf hi) {
    return __builtin_shufflevector(lo, hi, 0,1,2,3,4,5,6,7,8,9,10,11,12,13,14,15);
}
// A-matrix fragment (16x32 bf16): lane-half lh gets K = lh*8..lh*8+7 and 16+lh*8..
__device__ __forceinline__ v16bf loadFragA(const __bf16* rowp, int lh) {
    v8bf lo = *(const v8bf*)(rowp + lh * 8);
    v8bf hi = *(const v8bf*)(rowp + 16 + lh * 8);
    return combine16(lo, hi);
}
// B-matrix fragment (32x16 bf16), LDS holds [n][k]: lane-half lh gets K = lh*8..lh*8+7 and 16+lh*8..
__device__ __forceinline__ v16bf loadFragB(const __bf16* rowp, int lh) {
    v8bf lo = *(const v8bf*)(rowp + lh * 8);
    v8bf hi = *(const v8bf*)(rowp + 16 + lh * 8);
    return combine16(lo, hi);
}
__device__ __forceinline__ v8f wmma_bf16(v16bf a, v16bf b, v8f c) {
    return __builtin_amdgcn_wmma_f32_16x16x32_bf16(false, a, false, b, (short)0, c, false, false);
}

// ---------------------------------------------------------------------------
// GEMM: C[M][N] = A[M][K] (bf16) x B[N][K]^T (fp32 weights, converted in-flight)
// BM=128 BN=128 BK=32, 256 threads = 8 waves (4x2), each wave 32x64 -> 8 WMMA.
// A tiles are moved by the Tensor Data Mover (tensor_load_to_lds) with HW LDS
// padding (16 DW data + 4 DW pad == our 80B row stride), double buffered and
// synchronized with s_wait_tensorcnt. B tiles are software-pipelined through
// registers and converted fp32->bf16 with the native cvt (hides under WMMA).
// EPI: 0=store fp32, 1=store bf16, 2=store fp32 + residual
// ---------------------------------------------------------------------------
#define BM 128
#define BN 128
#define BK 32
#define LDSK 40   // +8 bf16 pad -> 80B row stride, conflict-free b128 frags

template <int EPI>
__global__ __launch_bounds__(256) void gemm_bf16_k(
    const __bf16* __restrict__ A, const float* __restrict__ B,
    void* __restrict__ Cout, const float* __restrict__ res,
    int M, int N, int K)
{
    __shared__ __bf16 aS[2][BM * LDSK];
    __shared__ __bf16 bS[2][BN * LDSK];

    const int tid  = threadIdx.x;
    const int lane = tid & 31;
    const int wave = tid >> 5;
    const int l15  = lane & 15;
    const int lh   = lane >> 4;
    const int mo   = (wave >> 1) * 32;   // wave M offset in tile
    const int no   = (wave & 1) * 64;    // wave N offset in tile
    const int m0   = blockIdx.x * BM;
    const int n0   = blockIdx.y * BN;
    const int r    = tid >> 1;           // tile row handled by this thread
    const int hf   = tid & 1;            // 16-element half

    v8f acc[2][4] = {};
    const int nk = K / BK;

    // ---- TDM: one descriptor per A tile (2D, pad 16DW+4DW => 80B rows) ----
    auto issueA = [&](int kb, int buf) {
        unsigned lds = (unsigned)(unsigned long long)(size_t)(&aS[buf][0]);
        unsigned long long ga =
            (unsigned long long)(size_t)(A + (size_t)m0 * K + (size_t)kb * BK);
        v4u g0;
        g0[0] = 1u;                                    // count=1, user mode
        g0[1] = lds;                                   // lds_addr
        g0[2] = (unsigned)(ga & 0xFFFFFFFFull);        // global_addr[31:0]
        g0[3] = (unsigned)((ga >> 32) & 0x01FFFFFFull) // global_addr[56:32]
              | (2u << 30);                            // type=2 (image/D#)
        v8i g1;
        g1[0] = (1 << 16)        // data_size = 2 bytes
              | (1 << 20)        // pad_enable
              | (3 << 22)        // pad_interval: 16 DWORDs
              | (3 << 25);       // pad_amount:   4 DWORDs
        g1[1] = (int)(((unsigned)K & 0xFFFFu) << 16);                      // tensor_dim0 lo16
        g1[2] = (int)((((unsigned)K >> 16) & 0xFFFFu)
              | (((unsigned)M & 0xFFFFu) << 16));                          // dim0 hi | dim1 lo
        g1[3] = (int)(((unsigned)M >> 16) & 0xFFFFu) | (BK << 16);         // dim1 hi | tile_dim0
        g1[4] = BM;                                                        // tile_dim1 (tile_dim2=0)
        g1[5] = K;                                                         // tensor_dim0_stride lo32
        g1[6] = 0;                                                         // stride hi / dim1_stride lo
        g1[7] = 0;
        v4i z = {0, 0, 0, 0};
#if defined(__clang_major__) && (__clang_major__ >= 23)
        v8i z8 = {0, 0, 0, 0, 0, 0, 0, 0};
        __builtin_amdgcn_tensor_load_to_lds(g0, g1, z, z, z8, 0);
#else
        __builtin_amdgcn_tensor_load_to_lds(g0, g1, z, z, 0);
#endif
    };

    // ---- B tile pipeline registers (loaded one k-step ahead) ----
    float br[16];
    auto loadB = [&](int kb) {
        int n = n0 + r;
        if (n < N) {
            const float* src = B + (size_t)n * K + kb * BK + hf * 16;
#pragma unroll
            for (int i = 0; i < 16; ++i) br[i] = src[i];
        } else {
#pragma unroll
            for (int i = 0; i < 16; ++i) br[i] = 0.f;
        }
    };

    if (wave == 0) issueA(0, 0);
    loadB(0);

    for (int kb = 0; kb < nk; ++kb) {
        const int cur = kb & 1;
        {   // convert + store B(kb) to LDS
            v8bf lo, hi;
#pragma unroll
            for (int i = 0; i < 8; ++i) { lo[i] = f2bf(br[i]); hi[i] = f2bf(br[8 + i]); }
            v8bf* dst = (v8bf*)(&bS[cur][r * LDSK + hf * 16]);
            dst[0] = lo;
            dst[1] = hi;
        }
        if (kb + 1 < nk) loadB(kb + 1);                       // prefetch next B
        if (wave == 0) __builtin_amdgcn_s_wait_tensorcnt(0);  // A(kb) landed
        __syncthreads();
        if (wave == 0 && kb + 1 < nk) issueA(kb + 1, cur ^ 1);// overlap next A DMA

        v16bf af[2], bf[4];
#pragma unroll
        for (int i = 0; i < 2; ++i)
            af[i] = loadFragA(&aS[cur][(mo + i * 16 + l15) * LDSK], lh);
#pragma unroll
        for (int j = 0; j < 4; ++j)
            bf[j] = loadFragB(&bS[cur][(no + j * 16 + l15) * LDSK], lh);
#pragma unroll
        for (int i = 0; i < 2; ++i)
#pragma unroll
            for (int j = 0; j < 4; ++j)
                acc[i][j] = wmma_bf16(af[i], bf[j], acc[i][j]);
    }

    // ---- epilogue: C layout row = vg + 8*lh, col = l15
#pragma unroll
    for (int i = 0; i < 2; ++i) {
#pragma unroll
        for (int j = 0; j < 4; ++j) {
            int col = n0 + no + j * 16 + l15;
            if (col >= N) continue;
            int rbase = m0 + mo + i * 16 + lh * 8;
#pragma unroll
            for (int vg = 0; vg < 8; ++vg) {
                float v = acc[i][j][vg];
                size_t idx = (size_t)(rbase + vg) * N + col;
                if (EPI == 0)      ((float*)Cout)[idx]  = v;
                else if (EPI == 1) ((__bf16*)Cout)[idx] = f2bf(v);
                else               ((float*)Cout)[idx]  = v + res[idx];
            }
        }
    }
}

// ---------------------------------------------------------------------------
// RMSNorm (fp32 in -> bf16 out), one block per row
// ---------------------------------------------------------------------------
__global__ __launch_bounds__(256) void rmsnorm_k(
    const float* __restrict__ in, const float* __restrict__ w,
    __bf16* __restrict__ out, int C, int inStride)
{
    __shared__ float red[256];
    const int row = blockIdx.x, tid = threadIdx.x;
    const float* x = in + (size_t)row * inStride;
    float ss = 0.f;
    for (int i = tid; i < C; i += 256) { float v = x[i]; ss += v * v; }
    red[tid] = ss;
    __syncthreads();
    for (int s = 128; s > 0; s >>= 1) {
        if (tid < s) red[tid] += red[tid + s];
        __syncthreads();
    }
    float scale = rsqrtf(red[0] / (float)C + 1e-6f);
    for (int i = tid; i < C; i += 256)
        out[(size_t)row * C + i] = f2bf(x[i] * scale * w[i]);
}

// ---------------------------------------------------------------------------
// q prep: q fp32 [T][H*192] -> q_full bf16 [H][T][192] with RoPE on last 64
// ---------------------------------------------------------------------------
__global__ void qprep_k(const float* __restrict__ q, const float* __restrict__ cs,
                        const float* __restrict__ sn, __bf16* __restrict__ qfull)
{
    int idx = blockIdx.x * 256 + threadIdx.x;
    if (idx >= NHEAD * T_SEQ * D_HEAD) return;
    int d = idx % D_HEAD;
    int t = (idx / D_HEAD) % T_SEQ;
    int h = idx / (D_HEAD * T_SEQ);
    const float* qr = q + (size_t)t * (NHEAD * D_HEAD) + h * D_HEAD;
    float v;
    if (d < D_NOPE) {
        v = qr[d];
    } else if (d < D_NOPE + 32) {
        int i = d - D_NOPE;
        v = qr[D_NOPE + 2 * i] * cs[t * 32 + i] - qr[D_NOPE + 2 * i + 1] * sn[t * 32 + i];
    } else {
        int i = d - D_NOPE - 32;
        v = qr[D_NOPE + 2 * i] * sn[t * 32 + i] + qr[D_NOPE + 2 * i + 1] * cs[t * 32 + i];
    }
    qfull[idx] = f2bf(v);
}

// k prep: kv bf16 [T][H*256] + ckv fp32 [T][576] -> k_full bf16 [H][T][192]
__global__ void kprep_k(const __bf16* __restrict__ kv, const float* __restrict__ ckv,
                        const float* __restrict__ cs, const float* __restrict__ sn,
                        __bf16* __restrict__ kfull)
{
    int idx = blockIdx.x * 256 + threadIdx.x;
    if (idx >= NHEAD * T_SEQ * D_HEAD) return;
    int d = idx % D_HEAD;
    int t = (idx / D_HEAD) % T_SEQ;
    int h = idx / (D_HEAD * T_SEQ);
    float v;
    if (d < D_NOPE) {
        v = bf2f(kv[(size_t)t * (NHEAD * 256) + h * 256 + d]);
    } else {
        const float* kp = ckv + (size_t)t * (KV_RANK + D_ROPE) + KV_RANK;
        if (d < D_NOPE + 32) {
            int i = d - D_NOPE;
            v = kp[2 * i] * cs[t * 32 + i] - kp[2 * i + 1] * sn[t * 32 + i];
        } else {
            int i = d - D_NOPE - 32;
            v = kp[2 * i] * sn[t * 32 + i] + kp[2 * i + 1] * cs[t * 32 + i];
        }
    }
    kfull[idx] = f2bf(v);
}

// v transpose: kv bf16 [T][H*256] (cols h*256+128..255) -> vT bf16 [H][128][T]
__global__ __launch_bounds__(256) void vtrans_k(const __bf16* __restrict__ kv,
                                                __bf16* __restrict__ vT)
{
    __shared__ __bf16 tile[64][68];
    const int h = blockIdx.x, t0 = blockIdx.y * 64, d0 = blockIdx.z * 64;
    const int tid = threadIdx.x;
    const int r = tid >> 2, q4 = tid & 3;
    const __bf16* src = kv + (size_t)(t0 + r) * (NHEAD * 256) + h * 256 + D_NOPE + d0 + q4 * 16;
    v8bf a = *(const v8bf*)src, b = *(const v8bf*)(src + 8);
#pragma unroll
    for (int i = 0; i < 8; ++i) { tile[r][q4 * 16 + i] = a[i]; tile[r][q4 * 16 + 8 + i] = b[i]; }
    __syncthreads();
    __bf16* dst = vT + ((size_t)h * D_V + d0 + r) * T_SEQ + t0 + q4 * 16;
    v8bf o1, o2;
#pragma unroll
    for (int i = 0; i < 8; ++i) { o1[i] = tile[q4 * 16 + i][r]; o2[i] = tile[q4 * 16 + 8 + i][r]; }
    *(v8bf*)dst = o1;
    *(v8bf*)(dst + 8) = o2;
}

// silu(gate)*up : gu bf16 [T][2*I_FF] -> act bf16 [T][I_FF]
__global__ void silu_k(const __bf16* __restrict__ gu, __bf16* __restrict__ act)
{
    int idx = blockIdx.x * 256 + threadIdx.x;
    if (idx >= T_SEQ * I_FF) return;
    int t = idx / I_FF, i = idx % I_FF;
    float g = bf2f(gu[(size_t)t * (2 * I_FF) + i]);
    float u = bf2f(gu[(size_t)t * (2 * I_FF) + I_FF + i]);
    float s = g / (1.f + __expf(-g));
    act[idx] = f2bf(s * u);
}

// ---------------------------------------------------------------------------
// Flash attention: grid (H, T/64), 128 threads = 4 waves, wave owns 16 q rows.
// ---------------------------------------------------------------------------
__global__ __launch_bounds__(128) void attn_k(
    const __bf16* __restrict__ qfull,  // [H][T][192]
    const __bf16* __restrict__ kfull,  // [H][T][192]
    const __bf16* __restrict__ vT,     // [H][128][T]
    __bf16* __restrict__ out)          // [T][H*128]
{
    __shared__ __bf16 kS[32 * 200];       // [s][192+8]
    __shared__ __bf16 vS[128 * 40];       // [d][32+8]
    __shared__ __bf16 pS[4 * 16 * 40];    // per-wave P tile [16][32+8]

    const int h = blockIdx.x, q0 = blockIdx.y * 64;
    const int tid = threadIdx.x, lane = tid & 31, wave = tid >> 5;
    const int l15 = lane & 15, lh = lane >> 4;

    v16bf qf[6];
    {
        const __bf16* qrow = qfull + ((size_t)h * T_SEQ + q0 + wave * 16 + l15) * D_HEAD;
#pragma unroll
        for (int kb = 0; kb < 6; ++kb) qf[kb] = loadFragA(qrow + kb * 32, lh);
    }

    v8f Oacc[8] = {};
    float mS[8], lS[8];
#pragma unroll
    for (int i = 0; i < 8; ++i) { mS[i] = -1e30f; lS[i] = 0.f; }

    const int qrow_base = q0 + wave * 16 + lh * 8;
    const int send = q0 + 64;
    for (int s0 = 0; s0 < send; s0 += 32) {
        __syncthreads();
        {   // K tile: 32 x 192 bf16
            int s = tid >> 2, c = (tid & 3) * 48;
            const __bf16* src = kfull + ((size_t)h * T_SEQ + s0 + s) * D_HEAD + c;
            __bf16* dst = kS + s * 200 + c;
#pragma unroll
            for (int u = 0; u < 6; ++u) *(v8bf*)(dst + u * 8) = *(const v8bf*)(src + u * 8);
        }
        {   // V tile (pre-transposed): 128 x 32 bf16
            int d = tid;
            const __bf16* src = vT + ((size_t)h * D_V + d) * T_SEQ + s0;
            __bf16* dst = vS + d * 40;
#pragma unroll
            for (int u = 0; u < 4; ++u) *(v8bf*)(dst + u * 8) = *(const v8bf*)(src + u * 8);
        }
        __syncthreads();

        v8f Sacc[2] = {};
#pragma unroll
        for (int nt = 0; nt < 2; ++nt)
#pragma unroll
            for (int kb = 0; kb < 6; ++kb) {
                v16bf b = loadFragB(kS + (nt * 16 + l15) * 200 + kb * 32, lh);
                Sacc[nt] = wmma_bf16(qf[kb], b, Sacc[nt]);
            }

        float rmax[8];
#pragma unroll
        for (int vg = 0; vg < 8; ++vg) rmax[vg] = -1e30f;
#pragma unroll
        for (int nt = 0; nt < 2; ++nt) {
            int scol = s0 + nt * 16 + l15;
#pragma unroll
            for (int vg = 0; vg < 8; ++vg) {
                float v = Sacc[nt][vg] * ATTN_SCALE;
                if (scol > qrow_base + vg) v = -1e30f;
                Sacc[nt][vg] = v;
                rmax[vg] = fmaxf(rmax[vg], v);
            }
        }
#pragma unroll
        for (int off = 1; off < 16; off <<= 1)
#pragma unroll
            for (int vg = 0; vg < 8; ++vg)
                rmax[vg] = fmaxf(rmax[vg], __shfl_xor(rmax[vg], off, 32));

        float alpha[8], rsum[8];
#pragma unroll
        for (int vg = 0; vg < 8; ++vg) {
            float mn = fmaxf(mS[vg], rmax[vg]);
            alpha[vg] = __expf(mS[vg] - mn);
            mS[vg] = mn;
            rsum[vg] = 0.f;
        }
#pragma unroll
        for (int nt = 0; nt < 2; ++nt)
#pragma unroll
            for (int vg = 0; vg < 8; ++vg) {
                float p = __expf(Sacc[nt][vg] - mS[vg]);
                Sacc[nt][vg] = p;
                rsum[vg] += p;
            }
#pragma unroll
        for (int off = 1; off < 16; off <<= 1)
#pragma unroll
            for (int vg = 0; vg < 8; ++vg)
                rsum[vg] += __shfl_xor(rsum[vg], off, 32);
#pragma unroll
        for (int vg = 0; vg < 8; ++vg) lS[vg] = lS[vg] * alpha[vg] + rsum[vg];
#pragma unroll
        for (int j = 0; j < 8; ++j)
#pragma unroll
            for (int vg = 0; vg < 8; ++vg) Oacc[j][vg] *= alpha[vg];

        __bf16* pw = pS + wave * 16 * 40;
#pragma unroll
        for (int nt = 0; nt < 2; ++nt)
#pragma unroll
            for (int vg = 0; vg < 8; ++vg)
                pw[(vg + 8 * lh) * 40 + nt * 16 + l15] = f2bf(Sacc[nt][vg]);
        v16bf pf = loadFragA(pw + l15 * 40, lh);   // same-wave DS ops are in-order

#pragma unroll
        for (int j = 0; j < 8; ++j) {
            v16bf b = loadFragB(vS + (j * 16 + l15) * 40, lh);
            Oacc[j] = wmma_bf16(pf, b, Oacc[j]);
        }
    }

#pragma unroll
    for (int j = 0; j < 8; ++j) {
        int col = j * 16 + l15;
#pragma unroll
        for (int vg = 0; vg < 8; ++vg) {
            int row = qrow_base + vg;
            out[(size_t)row * (NHEAD * D_V) + h * D_V + col] = f2bf(Oacc[j][vg] / lS[vg]);
        }
    }
}

// ---------------------------------------------------------------------------
// Host launcher
// ---------------------------------------------------------------------------
extern "C" void kernel_launch(void* const* d_in, const int* in_sizes, int n_in,
                              void* d_out, int out_size, void* d_ws, size_t ws_size,
                              hipStream_t stream)
{
    const float* hidden   = (const float*)d_in[0];
    const float* cosT     = (const float*)d_in[1];
    const float* sinT     = (const float*)d_in[2];
    const float* w_ln_in  = (const float*)d_in[3];
    const float* w_q_a    = (const float*)d_in[4];
    const float* w_q_ln   = (const float*)d_in[5];
    const float* w_q_b    = (const float*)d_in[6];
    const float* w_kv_a   = (const float*)d_in[7];
    const float* w_kv_ln  = (const float*)d_in[8];
    const float* w_kv_b   = (const float*)d_in[9];
    const float* w_o      = (const float*)d_in[10];
    const float* w_ln_post= (const float*)d_in[11];
    const float* w_gu     = (const float*)d_in[12];
    const float* w_down   = (const float*)d_in[13];
    float* outp = (float*)d_out;

    char* w = (char*)d_ws;
    auto alloc = [&](size_t bytes) -> char* {
        char* p = w;
        w += (bytes + 255) & ~(size_t)255;
        return p;
    };
    __bf16* x_bf    = (__bf16*)alloc((size_t)T_SEQ * DMODEL * 2);
    float*  qa_f    = (float*) alloc((size_t)T_SEQ * Q_RANK * 4);
    __bf16* qa_bf   = (__bf16*)alloc((size_t)T_SEQ * Q_RANK * 2);
    float*  ckv_f   = (float*) alloc((size_t)T_SEQ * (KV_RANK + D_ROPE) * 4);
    __bf16* c_bf    = (__bf16*)alloc((size_t)T_SEQ * KV_RANK * 2);
    float*  q_f     = (float*) alloc((size_t)T_SEQ * NHEAD * D_HEAD * 4);
    __bf16* kv_bf   = (__bf16*)alloc((size_t)T_SEQ * NHEAD * 256 * 2);
    __bf16* qfull   = (__bf16*)alloc((size_t)NHEAD * T_SEQ * D_HEAD * 2);
    __bf16* kfull   = (__bf16*)alloc((size_t)NHEAD * T_SEQ * D_HEAD * 2);
    __bf16* vT      = (__bf16*)alloc((size_t)NHEAD * D_V * T_SEQ * 2);
    __bf16* attn_bf = (__bf16*)alloc((size_t)T_SEQ * NHEAD * D_V * 2);
    float*  h_f     = (float*) alloc((size_t)T_SEQ * DMODEL * 4);
    __bf16* x2_bf   = (__bf16*)alloc((size_t)T_SEQ * DMODEL * 2);
    __bf16* gu_bf   = (__bf16*)alloc((size_t)T_SEQ * 2 * I_FF * 2);
    __bf16* act_bf  = (__bf16*)alloc((size_t)T_SEQ * I_FF * 2);

    auto gemmGrid = [](int M, int N) { return dim3(M / BM, (N + BN - 1) / BN); };

    rmsnorm_k<<<T_SEQ, 256, 0, stream>>>(hidden, w_ln_in, x_bf, DMODEL, DMODEL);
    gemm_bf16_k<0><<<gemmGrid(T_SEQ, Q_RANK), 256, 0, stream>>>(x_bf, w_q_a, qa_f, nullptr, T_SEQ, Q_RANK, DMODEL);
    gemm_bf16_k<0><<<gemmGrid(T_SEQ, KV_RANK + D_ROPE), 256, 0, stream>>>(x_bf, w_kv_a, ckv_f, nullptr, T_SEQ, KV_RANK + D_ROPE, DMODEL);
    rmsnorm_k<<<T_SEQ, 256, 0, stream>>>(qa_f, w_q_ln, qa_bf, Q_RANK, Q_RANK);
    rmsnorm_k<<<T_SEQ, 256, 0, stream>>>(ckv_f, w_kv_ln, c_bf, KV_RANK, KV_RANK + D_ROPE);
    gemm_bf16_k<0><<<gemmGrid(T_SEQ, NHEAD * D_HEAD), 256, 0, stream>>>(qa_bf, w_q_b, q_f, nullptr, T_SEQ, NHEAD * D_HEAD, Q_RANK);
    gemm_bf16_k<1><<<gemmGrid(T_SEQ, NHEAD * 256), 256, 0, stream>>>(c_bf, w_kv_b, kv_bf, nullptr, T_SEQ, NHEAD * 256, KV_RANK);
    {
        int tot = NHEAD * T_SEQ * D_HEAD;
        qprep_k<<<(tot + 255) / 256, 256, 0, stream>>>(q_f, cosT, sinT, qfull);
        kprep_k<<<(tot + 255) / 256, 256, 0, stream>>>(kv_bf, ckv_f, cosT, sinT, kfull);
        vtrans_k<<<dim3(NHEAD, T_SEQ / 64, D_V / 64), 256, 0, stream>>>(kv_bf, vT);
    }
    attn_k<<<dim3(NHEAD, T_SEQ / 64), 128, 0, stream>>>(qfull, kfull, vT, attn_bf);
    gemm_bf16_k<2><<<gemmGrid(T_SEQ, DMODEL), 256, 0, stream>>>(attn_bf, w_o, h_f, hidden, T_SEQ, DMODEL, NHEAD * D_V);
    rmsnorm_k<<<T_SEQ, 256, 0, stream>>>(h_f, w_ln_post, x2_bf, DMODEL, DMODEL);
    gemm_bf16_k<1><<<gemmGrid(T_SEQ, 2 * I_FF), 256, 0, stream>>>(x2_bf, w_gu, gu_bf, nullptr, T_SEQ, 2 * I_FF, DMODEL);
    silu_k<<<(T_SEQ * I_FF + 255) / 256, 256, 0, stream>>>(gu_bf, act_bf);
    gemm_bf16_k<2><<<gemmGrid(T_SEQ, DMODEL), 256, 0, stream>>>(act_bf, w_down, outp, h_f, T_SEQ, DMODEL, I_FF);

    (void)in_sizes; (void)n_in; (void)out_size; (void)ws_size;
}
